// ConvolutionLayer_19894288515508
// MI455X (gfx1250) — matrix operand order repaired and checked by reading, not compiled, not run on hardware
//
#include <hip/hip_runtime.h>
#include <hip/hip_bf16.h>

// ---------------------------------------------------------------------------
// Problem constants (from reference)
// ---------------------------------------------------------------------------
#define NN      50000      // nodes
#define NE      800000     // edges
#define FN      128        // node feature dim
#define FE      64         // edge feature dim
#define ZD      320        // 2*FN + FE
#define BN_EPS  1e-5f

// GEMM tiling
#define EPB        128         // edges per workgroup (M tile)
#define TPB        512         // threads per block (16 waves)
#define WT_ROWS    256         // combined [Wf;Ws]^T rows (N total)
#define PITCH_E    328         // LDS row pitch in bf16 elements (padded, 16B-mult)
#define PITCH_B    (PITCH_E*2) // 656 bytes

// LDS layout (bytes)
#define WT_OFF   0
#define WT_BYTES (WT_ROWS * PITCH_B)            // 167936
#define A_OFF    (WT_OFF + WT_BYTES)            // 167936
#define A_BYTES  (EPB * PITCH_B)                // 83968
#define DST_OFF  (A_OFF + A_BYTES)              // 251904
#define SRC_OFF  (DST_OFF + EPB * 4)            // 252416
#define LDS_BYTES (SRC_OFF + EPB * 4)           // 252928  (< 320KB / WGP)

// Workspace layout (bytes)
#define WS_WT1   0
#define WS_WT2   (WS_WT1 + WT_ROWS * ZD * 2)    // 163840
#define WS_CNT   (WS_WT2 + WT_ROWS * ZD * 2)    // 327680
#define WS_SUM   ((WS_CNT + NN * 4 + 255) & ~255ull)  // aligned
#define SUM_BYTES ((size_t)NN * FN * 4)

typedef __attribute__((ext_vector_type(8)))  float   v8f;
typedef __attribute__((ext_vector_type(16))) __bf16  v16bf;

// sched_group_barrier masks
#define SG_WMMA    0x008u   // MFMA/WMMA instructions
#define SG_DSREAD  0x100u   // DS read instructions

// float -> bf16 bits, round-to-nearest-even (manual, avoids __bf16 arith)
__device__ __forceinline__ unsigned short f2bf(float f) {
    union { float f; unsigned u; } v; v.f = f;
    unsigned u = v.u;
    u += 0x7FFFu + ((u >> 16) & 1u);
    return (unsigned short)(u >> 16);
}

// load a 32-byte WMMA fragment from two 16B LDS chunks (bit-cast to v16bf)
__device__ __forceinline__ v16bf ld_frag(const char* p0, const char* p1) {
    union { uint4 q[2]; v16bf v; } u;
    u.q[0] = *(const uint4*)p0;
    u.q[1] = *(const uint4*)p1;
    return u.v;
}

__device__ __forceinline__ void atomic_fadd(float* p, float v) {
    __hip_atomic_fetch_add(p, v, __ATOMIC_RELAXED, __HIP_MEMORY_SCOPE_AGENT);
}

// ---------------------------------------------------------------------------
// Prep: build combined transposed bf16 weights Wt[n][k] = (n<128?Wf:Ws)[k][n%128]
// for both layers. idx space: 2 * 256 * 320
// ---------------------------------------------------------------------------
__global__ void prep_weights(const float* __restrict__ Wf1, const float* __restrict__ Ws1,
                             const float* __restrict__ Wf2, const float* __restrict__ Ws2,
                             unsigned short* __restrict__ wt1,
                             unsigned short* __restrict__ wt2) {
    int idx = blockIdx.x * blockDim.x + threadIdx.x;      // 0 .. 163839
    int layer = idx / (WT_ROWS * ZD);
    int rem   = idx % (WT_ROWS * ZD);
    int n = rem / ZD, k = rem % ZD;
    const float* Wf = layer ? Wf2 : Wf1;
    const float* Ws = layer ? Ws2 : Ws1;
    float v = (n < FN) ? Wf[k * FN + n] : Ws[k * FN + (n - FN)];
    (layer ? wt2 : wt1)[n * ZD + k] = f2bf(v);
}

// ---------------------------------------------------------------------------
// Degree count (dst identical for both layers)
// ---------------------------------------------------------------------------
__global__ void count_deg(const int* __restrict__ ei, float* __restrict__ cnt) {
    int e = blockIdx.x * blockDim.x + threadIdx.x;
    if (e < NE) atomic_fadd(&cnt[ei[NE + e]], 1.0f);
}

// ---------------------------------------------------------------------------
// Edge GEMM + gated activation + scatter-add.
// 512 threads = 16 waves. Wave w: M-subtile (w>>1), column-block group (w&1):
// 4 blocks of 16 columns, for BOTH Wf and Ws halves (64 acc VGPRs/wave).
// ---------------------------------------------------------------------------
__global__ void __launch_bounds__(TPB, 1)
cgconv_gemm(const float* __restrict__ x, const int* __restrict__ ei,
            const float* __restrict__ ea, const unsigned short* __restrict__ wt,
            const float* __restrict__ biasF, const float* __restrict__ biasS,
            float* __restrict__ summed) {
    extern __shared__ char smem[];
    const int tid = threadIdx.x;
    const int e0  = blockIdx.x * EPB;

    int* sDst = (int*)(smem + DST_OFF);
    int* sSrc = (int*)(smem + SRC_OFF);
    if (tid < EPB) {
        sSrc[tid] = ei[e0 + tid];
        sDst[tid] = ei[NE + e0 + tid];
    }
    __syncthreads();

    // ---- stage combined bf16 weights [256][320] -> LDS (padded pitch) ----
    {
        const uint4* g = (const uint4*)wt;             // 40 x 16B chunks per row
        #pragma unroll 4
        for (int i = 0; i < 20; ++i) {
            int idx = tid + i * TPB;                   // 10240 chunks
            int n = idx / 40, k8 = idx % 40;
            *(uint4*)(smem + WT_OFF + n * PITCH_B + k8 * 16) = g[n * 40 + k8];
        }
    }
    // ---- stage A tile: z = [x[dst] | x[src] | edge_attr] as bf16 ----
    #pragma unroll 4
    for (int i = 0; i < 20; ++i) {
        int idx = tid + i * TPB;                       // 128 edges * 80 float4
        int el = idx / 80, q = idx % 80;
        const float4* gp;
        if (q < 32)        gp = (const float4*)(x + (size_t)sDst[el] * FN) + q;
        else if (q < 64)   gp = (const float4*)(x + (size_t)sSrc[el] * FN) + (q - 32);
        else               gp = (const float4*)(ea + (size_t)(e0 + el) * FE) + (q - 64);
        float4 v = *gp;
        uint2 p;
        p.x = (unsigned)f2bf(v.x) | ((unsigned)f2bf(v.y) << 16);
        p.y = (unsigned)f2bf(v.z) | ((unsigned)f2bf(v.w) << 16);
        *(uint2*)(smem + A_OFF + el * PITCH_B + q * 8) = p;
    }
    __syncthreads();

    const int wave  = tid >> 5;        // 0..15
    const int lane  = tid & 31;
    const int half  = lane >> 4;       // 0: lanes 0-15, 1: lanes 16-31
    const int l15   = lane & 15;
    const int msub  = wave >> 1;       // 0..7 : which 16-edge subtile
    const int ngrp  = wave & 1;        // 0..1 : which group of 4 column blocks

    v8f accF[4], accS[4];
    #pragma unroll
    for (int c = 0; c < 4; ++c) {
        accF[c] = (v8f){0,0,0,0,0,0,0,0};
        accS[c] = (v8f){0,0,0,0,0,0,0,0};
    }

    const char* aRow  = smem + A_OFF + (msub * 16 + l15) * PITCH_B;
    const char* wRowF = smem + WT_OFF + (ngrp * 64 + l15) * PITCH_B;   // first F block row
    const char* wRowS = wRowF + 128 * PITCH_B;                         // matching S block row

    #pragma unroll
    for (int ks = 0; ks < 10; ++ks) {
        const int k0 = ks * 32;
        // A frag (16-bit 16x32 layout): K chunks [kb0..kb0+8) and [kb0+16..kb0+24)
        const int kb0 = k0 + half * 8;
        v16bf af = ld_frag(aRow + kb0 * 2, aRow + kb0 * 2 + 32);
        // B frags: same K chunks as A ([kb0..kb0+8) and [kb0+16..kb0+24)) of column (lane&15)
        const int kbb = (k0 + half * 8) * 2;
        v16bf bF[4], bS[4];
        #pragma unroll
        for (int c = 0; c < 4; ++c) {
            const char* rF = wRowF + c * 16 * PITCH_B + kbb;
            const char* rS = wRowS + c * 16 * PITCH_B + kbb;
            bF[c] = ld_frag(rF, rF + 32);
            bS[c] = ld_frag(rS, rS + 32);
        }
        #pragma unroll
        for (int c = 0; c < 4; ++c) {
            accF[c] = __builtin_amdgcn_wmma_f32_16x16x32_bf16(
                false, af, false, bF[c], (short)0, accF[c], false, false);
            accS[c] = __builtin_amdgcn_wmma_f32_16x16x32_bf16(
                false, af, false, bS[c], (short)0, accS[c], false, false);
        }
        // Software-pipeline the DS reads one fragment-group ahead of the WMMAs:
        //   DS x6 (A + g0), DS x4 (g1), WMMA x2 (g0), DS x4 (g2), WMMA x2 (g1),
        //   DS x4 (g3), WMMA x2 (g2), WMMA x2 (g3 -> overlapped by next step's DS x6)
        __builtin_amdgcn_sched_group_barrier(SG_DSREAD, 6, 0);
        __builtin_amdgcn_sched_group_barrier(SG_DSREAD, 4, 0);
        __builtin_amdgcn_sched_group_barrier(SG_WMMA,   2, 0);
        __builtin_amdgcn_sched_group_barrier(SG_DSREAD, 4, 0);
        __builtin_amdgcn_sched_group_barrier(SG_WMMA,   2, 0);
        __builtin_amdgcn_sched_group_barrier(SG_DSREAD, 4, 0);
        __builtin_amdgcn_sched_group_barrier(SG_WMMA,   2, 0);
        __builtin_amdgcn_sched_group_barrier(SG_WMMA,   2, 0);
    }

    // ---- epilogue: msg = sigmoid(f+bf) * softplus(s+bs); scatter to summed ----
    #pragma unroll
    for (int c = 0; c < 4; ++c) {
        const int col = (ngrp * 4 + c) * 16 + l15;
        const float bfc = biasF[col];
        const float bsc = biasS[col];
        #pragma unroll
        for (int r = 0; r < 8; ++r) {
            const int rowLocal = msub * 16 + half * 8 + r;  // C/D layout: M = r (+8 hi lanes)
            const int d = sDst[rowLocal];
            float f = accF[c][r] + bfc;
            float s = accS[c][r] + bsc;
            float sig = 1.0f / (1.0f + __expf(-f));
            float sp  = (s > 20.0f) ? s : __logf(1.0f + __expf(s));
            atomic_fadd(&summed[(size_t)d * FN + col], sig * sp);
        }
    }
}

// ---------------------------------------------------------------------------
// Finalize: mean aggregate + BatchNorm(eval) + residual
// ---------------------------------------------------------------------------
__global__ void finalize(const float* __restrict__ summed, const float* __restrict__ cnt,
                         const float* __restrict__ gamma, const float* __restrict__ beta,
                         const float* __restrict__ rmean, const float* __restrict__ rvar,
                         const float* __restrict__ xin, float* __restrict__ out) {
    int idx = blockIdx.x * blockDim.x + threadIdx.x;     // NN*FN
    int n = idx >> 7, c = idx & 127;
    if (n >= NN) return;
    float denom = fmaxf(cnt[n], 1.0f);
    float v = summed[idx] / denom;
    v = (v - rmean[c]) * rsqrtf(rvar[c] + BN_EPS) * gamma[c] + beta[c];
    out[idx] = v + xin[idx];
}

// ---------------------------------------------------------------------------
// Host launcher
// ---------------------------------------------------------------------------
extern "C" void kernel_launch(void* const* d_in, const int* in_sizes, int n_in,
                              void* d_out, int out_size, void* d_ws, size_t ws_size,
                              hipStream_t stream) {
    (void)in_sizes; (void)n_in; (void)out_size; (void)ws_size;

    const float* x   = (const float*)d_in[0];
    const int*   ei  = (const int*)d_in[1];
    const float* ea  = (const float*)d_in[2];
    const float* Wf1 = (const float*)d_in[3];
    const float* bf1 = (const float*)d_in[4];
    const float* Ws1 = (const float*)d_in[5];
    const float* bs1 = (const float*)d_in[6];
    const float* g1  = (const float*)d_in[7];
    const float* be1 = (const float*)d_in[8];
    const float* rm1 = (const float*)d_in[9];
    const float* rv1 = (const float*)d_in[10];
    const float* Wf2 = (const float*)d_in[11];
    const float* bf2 = (const float*)d_in[12];
    const float* Ws2 = (const float*)d_in[13];
    const float* bs2 = (const float*)d_in[14];
    const float* g2  = (const float*)d_in[15];
    const float* be2 = (const float*)d_in[16];
    const float* rm2 = (const float*)d_in[17];
    const float* rv2 = (const float*)d_in[18];
    float* out = (float*)d_out;

    char* ws = (char*)d_ws;
    unsigned short* wt1 = (unsigned short*)(ws + WS_WT1);
    unsigned short* wt2 = (unsigned short*)(ws + WS_WT2);
    float* cnt    = (float*)(ws + WS_CNT);
    float* summed = (float*)(ws + WS_SUM);

    // allow >64KB dynamic LDS for the GEMM kernel (idempotent, capture-safe)
    hipFuncSetAttribute((const void*)cgconv_gemm,
                        hipFuncAttributeMaxDynamicSharedMemorySize, LDS_BYTES);

    hipMemsetAsync(cnt, 0, (size_t)NN * 4, stream);
    hipMemsetAsync(summed, 0, SUM_BYTES, stream);

    prep_weights<<<(2 * WT_ROWS * ZD) / 256, 256, 0, stream>>>(Wf1, Ws1, Wf2, Ws2, wt1, wt2);
    count_deg<<<(NE + 255) / 256, 256, 0, stream>>>(ei, cnt);

    // ---- layer 1 ----
    cgconv_gemm<<<NE / EPB, TPB, LDS_BYTES, stream>>>(x, ei, ea, wt1, bf1, bs1, summed);
    finalize<<<(NN * FN) / 256, 256, 0, stream>>>(summed, cnt, g1, be1, rm1, rv1, x, out);

    // ---- layer 2 (reads layer-1 output from d_out, updates d_out in place) ----
    hipMemsetAsync(summed, 0, SUM_BYTES, stream);
    cgconv_gemm<<<NE / EPB, TPB, LDS_BYTES, stream>>>(out, ei, ea, wt2, bf2, bs2, summed);
    finalize<<<(NN * FN) / 256, 256, 0, stream>>>(summed, cnt, g2, be2, rm2, rv2, out, out);
}
